// WrappedVisionAttention_57166014709774
// MI455X (gfx1250) — matrix operand order repaired and checked by reading, not compiled, not run on hardware
//
#include <hip/hip_runtime.h>
#include <hip/hip_bf16.h>

// ---------------------------------------------------------------------------
// Vision attention for MI455X / gfx1250 (wave32, WMMA 16x16x32 f16, fp32 acc)
// S=2304, H=1280, NH=16, HD=80 (padded to 96 for K-dim of QK^T WMMA)
// GEMMs and flash attention: async global->LDS double-buffered staging
// (global_load_async_to_lds_b128 / ASYNCcnt pipeline).
// ---------------------------------------------------------------------------

#define S_   2304
#define H_   1280
#define NH_  16
#define HD_  80
#define HDP_ 96
#define H3_  3840

typedef __attribute__((ext_vector_type(16))) _Float16 v16h;
typedef __attribute__((ext_vector_type(8)))  _Float16 v8h;
typedef __attribute__((ext_vector_type(8)))  float    v8f;

union F16x16 { v16h v; v8h h[2]; _Float16 e[16]; };

#define LDS_PAD 40   // 32 data halves + 8 pad -> bank step 20, conflict-free
#define KPAD    104  // 96 data halves + 8 pad -> bank step 52, conflict-free

// ---- global-memory fragment loaders ---------------------------------------
// A operand (16x32 f16, ISA 7.12.2 interleaved):
// lanes 0-15: rows r0.., K {kb+0..7, kb+16..23}; lanes 16-31: K {kb+8..15, kb+24..31}
__device__ __forceinline__ v16h ldA(const _Float16* __restrict__ p, int ld, int r0, int kb, int lane) {
    const _Float16* q = p + (size_t)(r0 + (lane & 15)) * ld + kb + ((lane >> 4) << 3);
    F16x16 f;
    f.h[0] = *(const v8h*)(q);
    f.h[1] = *(const v8h*)(q + 16);
    return f.v;
}
// B operand (32x16 f16): lane n = column n0+(n&15); lanes 0-15 K {kb+0..7, kb+16..23},
// lanes 16-31 K {kb+8..15, kb+24..31} (interleaved, same as A)
__device__ __forceinline__ v16h ldB(const _Float16* __restrict__ p, int ld, int n0, int kb, int lane) {
    const _Float16* q = p + (size_t)(n0 + (lane & 15)) * ld + kb + ((lane >> 4) << 3);
    F16x16 f;
    f.h[0] = *(const v8h*)(q);
    f.h[1] = *(const v8h*)(q + 16);
    return f.v;
}

// ---- LDS fragment loaders -------------------------------------------------
__device__ __forceinline__ v16h ldAs(const _Float16 (*s)[LDS_PAD], int r0, int lane) {
    const _Float16* q = &s[r0 + (lane & 15)][(lane >> 4) << 3];
    F16x16 f;
    f.h[0] = *(const v8h*)(q);
    f.h[1] = *(const v8h*)(q + 16);
    return f.v;
}
__device__ __forceinline__ v16h ldBs(const _Float16 (*s)[LDS_PAD], int n0, int lane) {
    const _Float16* q = &s[n0 + (lane & 15)][(lane >> 4) << 3];
    F16x16 f;
    f.h[0] = *(const v8h*)(q);
    f.h[1] = *(const v8h*)(q + 16);
    return f.v;
}
// K-tile A-operand loader from LDS (rows = keys, 96 dims padded to 104)
__device__ __forceinline__ v16h ldKs(const _Float16 (*s)[KPAD], int r0, int kb, int lane) {
    const _Float16* q = &s[r0 + (lane & 15)][kb + ((lane >> 4) << 3)];
    F16x16 f;
    f.h[0] = *(const v8h*)(q);
    f.h[1] = *(const v8h*)(q + 16);
    return f.v;
}

__device__ __forceinline__ v8f wmma16(v16h a, v16h b, v8f c) {
    return __builtin_amdgcn_wmma_f32_16x16x32_f16(false, a, false, b, (short)0, c, false, false);
}

// ---- CDNA5 async global->LDS copy (tracked by ASYNCcnt) -------------------
__device__ __forceinline__ void async_b128(unsigned lds_off, const _Float16* g) {
    asm volatile("global_load_async_to_lds_b128 %0, %1, off"
                 :: "v"(lds_off), "v"((unsigned long long)(size_t)g)
                 : "memory");
}
#define WAIT_ASYNC(n) asm volatile("s_wait_asynccnt %0" :: "n"(n) : "memory")

// ---------------------------------------------------------------------------
// fp32 -> f16 convert
// ---------------------------------------------------------------------------
__global__ void __launch_bounds__(256) cvt_f32_f16(const float* __restrict__ in,
                                                   _Float16* __restrict__ out, int n) {
    int i = blockIdx.x * blockDim.x + threadIdx.x;
    if (i < n) out[i] = (_Float16)in[i];
}

// ---------------------------------------------------------------------------
// NT GEMM: C[M][N] = A[M][K] * B[N][K]^T + bias[N], f16 in, fp32 out.
// Block = 8 waves -> 128x128 tile; K-slices (128x32 of A and B) staged into
// double-buffered LDS with global_load_async_to_lds_b128; each wave computes
// a 32x64 sub-tile (8 WMMA per K-step) from LDS.
// Requires M%128==0, N%128==0, K%32==0 (true for all call sites).
// ---------------------------------------------------------------------------
__global__ void __launch_bounds__(256) gemm_nt_wmma(
    const _Float16* __restrict__ A, const _Float16* __restrict__ B,
    const float* __restrict__ bias, float* __restrict__ C,
    int M, int N, int K) {
    __shared__ _Float16 sA[2][128][LDS_PAD];
    __shared__ _Float16 sB[2][128][LDS_PAD];

    const int tid  = threadIdx.x;
    const int lane = tid & 31;
    const int wv   = tid >> 5;            // 0..7
    const int wm   = (wv >> 1) << 5;      // 0,32,64,96
    const int wn   = (wv & 1) << 6;       // 0,64
    const int tilesN = N >> 7;
    const int bm = (blockIdx.x / tilesN) << 7;
    const int bn = (blockIdx.x % tilesN) << 7;

    // cooperative copy: 1024 16B chunks per buffer, 4 per thread
    const int cr = tid >> 2;              // row 0..63 (and +64)
    const int sg = (tid & 3) << 3;        // K-seg in halves: 0,8,16,24
    const _Float16* gA1 = A + (size_t)(bm + cr) * K + sg;
    const _Float16* gA2 = A + (size_t)(bm + cr + 64) * K + sg;
    const _Float16* gB1 = B + (size_t)(bn + cr) * K + sg;
    const _Float16* gB2 = B + (size_t)(bn + cr + 64) * K + sg;
    unsigned lA1[2], lA2[2], lB1[2], lB2[2];
#pragma unroll
    for (int b = 0; b < 2; ++b) {
        lA1[b] = (unsigned)(size_t)&sA[b][cr][sg];
        lA2[b] = (unsigned)(size_t)&sA[b][cr + 64][sg];
        lB1[b] = (unsigned)(size_t)&sB[b][cr][sg];
        lB2[b] = (unsigned)(size_t)&sB[b][cr + 64][sg];
    }

    // prologue: stage K-slice 0 into buffer 0
    async_b128(lA1[0], gA1);
    async_b128(lA2[0], gA2);
    async_b128(lB1[0], gB1);
    async_b128(lB2[0], gB2);

    v8f acc[2][4] = {};
    const int ksteps = K >> 5;
    for (int ks = 0; ks < ksteps; ++ks) {
        const int cur = ks & 1;
        if (ks + 1 < ksteps) {
            const int nk = (ks + 1) << 5;
            const int nb = cur ^ 1;
            async_b128(lA1[nb], gA1 + nk);
            async_b128(lA2[nb], gA2 + nk);
            async_b128(lB1[nb], gB1 + nk);
            async_b128(lB2[nb], gB2 + nk);
            WAIT_ASYNC(4);   // async loads complete in order: old buffer done
        } else {
            WAIT_ASYNC(0);
        }
        __syncthreads();

        v16h a0 = ldAs(sA[cur], wm, lane);
        v16h a1 = ldAs(sA[cur], wm + 16, lane);
#pragma unroll
        for (int j = 0; j < 4; ++j) {
            v16h b = ldBs(sB[cur], wn + 16 * j, lane);
            acc[0][j] = wmma16(a0, b, acc[0][j]);
            acc[1][j] = wmma16(a1, b, acc[1][j]);
        }
        __syncthreads();     // all waves done reading before buffer reuse
    }

    const int r0 = bm + wm, c0 = bn + wn;
    const int hi8 = (lane >> 4) << 3;
#pragma unroll
    for (int i = 0; i < 2; ++i)
#pragma unroll
        for (int j = 0; j < 4; ++j) {
            const int col = c0 + 16 * j + (lane & 15);
            const float bv = bias ? bias[col] : 0.0f;
#pragma unroll
            for (int t = 0; t < 8; ++t) {
                const int row = r0 + 16 * i + t + hi8;
                C[(size_t)row * N + col] = acc[i][j][t] + bv;
            }
        }
}

// ---------------------------------------------------------------------------
// RoPE + per-head scatter.
//  qh[head][s][96] f16 : rope(q) * (log2e / sqrt(HD))  (softmax scale folded)
//  kh[head][s][96] f16 : rope(k)            dims 80..95 zero padded
//  vt[head][dim][s] f16: V transposed, dims padded to 96 (rows 80..95 zero)
// ---------------------------------------------------------------------------
__global__ void __launch_bounds__(256) rope_scatter(
    const float* __restrict__ qkv, const float* __restrict__ cosb,
    const float* __restrict__ sinb, _Float16* __restrict__ qh,
    _Float16* __restrict__ kh, _Float16* __restrict__ vt) {
    int idx = blockIdx.x * blockDim.x + threadIdx.x;
    const int total = S_ * NH_ * HDP_;
    if (idx >= total) return;
    const int d = idx % HDP_;
    const int r = idx / HDP_;
    const int head = r % NH_;
    const int s = r / NH_;
    const size_t qo = ((size_t)head * S_ + s) * HDP_ + d;
    const size_t vo = ((size_t)head * HDP_ + d) * S_ + s;
    if (d >= HD_) {
        qh[qo] = (_Float16)0.0f;
        kh[qo] = (_Float16)0.0f;
        vt[vo] = (_Float16)0.0f;
        return;
    }

    const float c  = cosb[s * HD_ + d];
    const float sn = sinb[s * HD_ + d];
    const size_t base = (size_t)s * H3_ + head * HD_;
    const float qv = qkv[base + d];
    const float kv = qkv[base + H_ + d];
    const float vv = qkv[base + 2 * H_ + d];
    const int   dr  = (d < 40) ? d + 40 : d - 40;
    const float sgn = (d < 40) ? -1.0f : 1.0f;
    const float qr = sgn * qkv[base + dr];
    const float kr = sgn * qkv[base + H_ + dr];
    const float QSC = 1.4426950408889634f / 8.944271909999159f;  // log2(e)/sqrt(80)

    qh[qo] = (_Float16)((qv * c + qr * sn) * QSC);
    kh[qo] = (_Float16)(kv * c + kr * sn);
    vt[vo] = (_Float16)vv;
}

// ---------------------------------------------------------------------------
// Flash attention. Block = 8 waves, all on the SAME head (144 q-tiles/head,
// 18 blocks/head). K (32x96) and V (96x32) chunk tiles staged into double-
// buffered LDS with async copies (768 16B chunks per buffer = 3 per thread),
// shared by all 8 waves -> 8x less L2 traffic.
// Scores computed transposed (S^T = K * Q^T) so softmax stats are per-lane and
// the exp'd P tile is already in the interleaved A-fragment layout for P*V.
// Single segment (cu_seqlens=[0,S]) -> mask is identically zero.
// ---------------------------------------------------------------------------
__global__ void __launch_bounds__(256) flash_attn_wmma(
    const _Float16* __restrict__ qh, const _Float16* __restrict__ kh,
    const _Float16* __restrict__ vt, _Float16* __restrict__ ao) {
    __shared__ _Float16 sK[2][32][KPAD];
    __shared__ _Float16 sV[2][96][LDS_PAD];

    const int tid  = threadIdx.x;
    const int lane = tid & 31;
    const int wv   = tid >> 5;
    const int head = blockIdx.x / 18;              // 18 blocks per head
    const int qt   = (blockIdx.x % 18) * 8 + wv;   // q-tile of this wave

    const _Float16* Q  = qh + (size_t)head * S_ * HDP_;
    const _Float16* Km = kh + (size_t)head * S_ * HDP_;
    const _Float16* Vp = vt + (size_t)head * HDP_ * S_;
    const int q0  = qt * 16;
    const int hi8 = (lane >> 4) << 3;

    // cooperative async copy mapping: 3 chunks per thread per buffer
    const _Float16* gp[3];
    int gstep[3];
    unsigned lo[3][2];
#pragma unroll
    for (int j = 0; j < 3; ++j) {
        const int id = tid + j * 256;
        if (id < 384) {                       // K tile: 32 rows x 12 chunks
            const int row = id / 12, seg = (id % 12) << 3;
            gp[j] = Km + (size_t)row * HDP_ + seg;
            gstep[j] = 32 * HDP_;
            lo[j][0] = (unsigned)(size_t)&sK[0][row][seg];
            lo[j][1] = (unsigned)(size_t)&sK[1][row][seg];
        } else {                              // V tile: 96 rows x 4 chunks
            const int vid = id - 384;
            const int row = vid >> 2, seg = (vid & 3) << 3;
            gp[j] = Vp + (size_t)row * S_ + seg;
            gstep[j] = 32;
            lo[j][0] = (unsigned)(size_t)&sV[0][row][seg];
            lo[j][1] = (unsigned)(size_t)&sV[1][row][seg];
        }
    }
    // prologue: stage chunk 0 into buffer 0
#pragma unroll
    for (int j = 0; j < 3; ++j) { async_b128(lo[j][0], gp[j]); gp[j] += gstep[j]; }

    v16h qf[3];
#pragma unroll
    for (int t = 0; t < 3; ++t) qf[t] = ldB(Q, HDP_, q0, t * 32, lane);  // Q^T as B operand

    v8f o[5] = {};
    float m = -3.0e38f, lsum = 0.0f;
    const int nchunks = S_ / 32;  // 72

    for (int ks = 0; ks < nchunks; ++ks) {
        const int cur = ks & 1;
        if (ks + 1 < nchunks) {
            const int nb = cur ^ 1;
#pragma unroll
            for (int j = 0; j < 3; ++j) { async_b128(lo[j][nb], gp[j]); gp[j] += gstep[j]; }
            WAIT_ASYNC(3);   // in-order completion: previous buffer done
        } else {
            WAIT_ASYNC(0);
        }
        __syncthreads();

        // ---- S^T tiles: rows = keys (local 0..31), cols = queries ----
        v8f s0 = {}, s1 = {};
#pragma unroll
        for (int t = 0; t < 3; ++t) {
            v16h k0 = ldKs(sK[cur], 0,  t * 32, lane);
            v16h k1 = ldKs(sK[cur], 16, t * 32, lane);
            s0 = wmma16(k0, qf[t], s0);
            s1 = wmma16(k1, qf[t], s1);
        }
        // ---- online softmax (per-query stats per-lane; combine halves) ----
        float mx = s0[0];
#pragma unroll
        for (int t = 1; t < 8; ++t) mx = fmaxf(mx, s0[t]);
#pragma unroll
        for (int t = 0; t < 8; ++t) mx = fmaxf(mx, s1[t]);
        mx = fmaxf(mx, __shfl_xor(mx, 16, 32));
        const float mnew = fmaxf(m, mx);

        float ls = 0.0f;
        F16x16 p;
#pragma unroll
        for (int t = 0; t < 8; ++t) { float e = exp2f(s0[t] - mnew); ls += e; p.e[t]     = (_Float16)e; }
#pragma unroll
        for (int t = 0; t < 8; ++t) { float e = exp2f(s1[t] - mnew); ls += e; p.e[8 + t] = (_Float16)e; }
        ls += __shfl_xor(ls, 16, 32);

        const float alpha = exp2f(m - mnew);
        lsum = lsum * alpha + ls;
        m = mnew;

        // broadcast per-query alpha into O's C-layout rows (row = j + hi8)
        float arow[8];
#pragma unroll
        for (int j = 0; j < 8; ++j) arow[j] = __shfl(alpha, j + hi8, 32);

        // ---- O = O*alpha + P (A-frag, no lane shuffles needed) * V ----
#pragma unroll
        for (int t = 0; t < 5; ++t) {
            v16h vf = ldBs(sV[cur], t * 16, lane);   // V^T rows = output dims
#pragma unroll
            for (int j = 0; j < 8; ++j) o[t][j] *= arow[j];
            o[t] = wmma16(p.v, vf, o[t]);
        }
        __syncthreads();     // all waves done reading before buffer reuse
    }

    float lrow[8];
#pragma unroll
    for (int j = 0; j < 8; ++j) lrow[j] = __shfl(lsum, j + hi8, 32);
#pragma unroll
    for (int t = 0; t < 5; ++t)
#pragma unroll
        for (int j = 0; j < 8; ++j) {
            const int row = q0 + j + hi8;
            const int col = head * HD_ + t * 16 + (lane & 15);
            ao[(size_t)row * H_ + col] = (_Float16)(o[t][j] / lrow[j]);
        }
}

// ---------------------------------------------------------------------------
// Host-side launch
// ---------------------------------------------------------------------------
extern "C" void kernel_launch(void* const* d_in, const int* in_sizes, int n_in,
                              void* d_out, int out_size, void* d_ws, size_t ws_size,
                              hipStream_t stream) {
    const float* hidden  = (const float*)d_in[0];   // S x H
    const float* w_qkv   = (const float*)d_in[1];   // 3H x H
    const float* b_qkv   = (const float*)d_in[2];   // 3H
    const float* w_proj  = (const float*)d_in[3];   // H x H
    const float* b_proj  = (const float*)d_in[4];   // H
    const float* cosb    = (const float*)d_in[5];   // S x HD
    const float* sinb    = (const float*)d_in[6];   // S x HD
    float*       out     = (float*)d_out;           // S x H fp32

    // workspace carve-out
    char* ws = (char*)d_ws;
    size_t off = 0;
    auto take = [&](size_t bytes) -> void* {
        void* p = ws + off;
        off = (off + bytes + 255) & ~(size_t)255;
        return p;
    };
    _Float16* hidden_h = (_Float16*)take((size_t)S_ * H_ * 2);
    _Float16* w_qkv_h  = (_Float16*)take((size_t)H3_ * H_ * 2);
    _Float16* w_proj_h = (_Float16*)take((size_t)H_ * H_ * 2);
    float*    qkv      = (float*)   take((size_t)S_ * H3_ * 4);
    _Float16* qh       = (_Float16*)take((size_t)NH_ * S_ * HDP_ * 2);
    _Float16* kh       = (_Float16*)take((size_t)NH_ * S_ * HDP_ * 2);
    _Float16* vt       = (_Float16*)take((size_t)NH_ * HDP_ * S_ * 2);
    _Float16* ao       = (_Float16*)take((size_t)S_ * H_ * 2);

    auto cdiv = [](int a, int b) { return (a + b - 1) / b; };

    // 1) precision convert
    cvt_f32_f16<<<cdiv(S_ * H_, 256), 256, 0, stream>>>(hidden, hidden_h, S_ * H_);
    cvt_f32_f16<<<cdiv(H3_ * H_, 256), 256, 0, stream>>>(w_qkv, w_qkv_h, H3_ * H_);
    cvt_f32_f16<<<cdiv(H_ * H_, 256), 256, 0, stream>>>(w_proj, w_proj_h, H_ * H_);

    // 2) QKV GEMM: (S x H) * (3H x H)^T -> S x 3H   [blocks = 18*30 = 540]
    gemm_nt_wmma<<<(S_ / 128) * (H3_ / 128), 256, 0, stream>>>(
        hidden_h, w_qkv_h, b_qkv, qkv, S_, H3_, H_);

    // 3) RoPE + per-head layout (pad HD->96, fold softmax scale into q)
    rope_scatter<<<cdiv(S_ * NH_ * HDP_, 256), 256, 0, stream>>>(
        qkv, cosb, sinb, qh, kh, vt);

    // 4) flash attention [16 heads x 18 blocks = 288 blocks, 8 waves each]
    flash_attn_wmma<<<NH_ * 18, 256, 0, stream>>>(qh, kh, vt, ao);

    // 5) output projection: (S x H) * (H x H)^T + b -> fp32 out [blocks = 18*10]
    gemm_nt_wmma<<<(S_ / 128) * (H_ / 128), 256, 0, stream>>>(
        ao, w_proj_h, b_proj, out, S_, H_, H_);
}
